// DynamicGraphPruning_45990509806148
// MI455X (gfx1250) — matrix operand order repaired and checked by reading, not compiled, not run on hardware
//
#include <hip/hip_runtime.h>
#include <hip/hip_bf16.h>

// ---------------------------------------------------------------------------
// DynamicGraphPruning for MI455X (gfx1250, wave32, WMMA)
//   features (8,16384,256) f32 -> proj GEMM (C=256 -> P=128) via bf16 WMMA
//   -> LayerNorm -> L2 normalize -> per-edge cosine sim + spatial sim.
// Workspace layout: [0,64KB): W^T bf16 (P x C); [64KB, +64MB): sem f32.
// ---------------------------------------------------------------------------

#define BN   8
#define LN_  16384          // H*W = 128*128
#define CN   256
#define PN   128
#define EN   65024          // 2*(H-1)*W + 2*H*(W-1)
#define LNEPS 1e-5f

typedef __attribute__((ext_vector_type(16))) __bf16 v16bf;
typedef __attribute__((ext_vector_type(8)))  __bf16 v8bf;
typedef __attribute__((ext_vector_type(8)))  float  v8f;

// ---- kernel 0: repack W_proj (C x P, f32, row-major) -> Wt (P x C, bf16) ----
__global__ __launch_bounds__(256) void dgp_prep_w(const float* __restrict__ Wp,
                                                  __bf16* __restrict__ Wt) {
    int i = blockIdx.x * 256 + threadIdx.x;       // over P*C
    if (i < PN * CN) {
        int p = i / CN, c = i % CN;
        Wt[i] = (__bf16)Wp[(size_t)c * PN + p];
    }
}

// ---- kernel 1: fused projection GEMM + bias + LayerNorm + L2 normalize -----
// One block = 128 threads (4 waves) = one 16-row tile of (B*L, P).
// Each wave computes a 16x32 slab via v_wmma_f32_16x16x32_bf16.
__global__ __launch_bounds__(128) void dgp_proj_ln(const float*  __restrict__ feat,
                                                   const __bf16* __restrict__ Wt,
                                                   const float*  __restrict__ bias,
                                                   const float*  __restrict__ gamma,
                                                   const float*  __restrict__ beta,
                                                   float*        __restrict__ sem) {
    __shared__ float hbuf[16][PN + 4];            // +4 pad kills bank conflicts

    const int blk  = blockIdx.x;                  // B * (L/16) blocks
    const int b    = blk >> 10;                   // L/16 = 1024
    const int row0 = (blk & 1023) << 4;
    const int tid  = threadIdx.x;
    const int wave = tid >> 5;
    const int lane = tid & 31;

    // --- WMMA phase ---------------------------------------------------------
    const int m   = lane & 15;
    const int kka = (lane < 16) ? 0 : 8;          // A-frag K interleave
    const int kkb = (lane < 16) ? 0 : 8;          // B-frag K interleave
    const float*  arow = feat + ((size_t)b * LN_ + row0 + m) * CN;
    const __bf16* w0   = Wt + (size_t)(wave * 32      + m) * CN; // N-tile 0 col
    const __bf16* w1   = Wt + (size_t)(wave * 32 + 16 + m) * CN; // N-tile 1 col

    v8f acc0 = {}, acc1 = {};

    #pragma unroll
    for (int kt = 0; kt < 8; ++kt) {
        const int k0 = kt * 32;

        // A fragment: 16x32 bf16, lane m holds K {kka..kka+7, kka+16..kka+23}
        v16bf a;
        {
            const float4* p = (const float4*)(arow + k0 + kka);
            float4 f0 = p[0], f1 = p[1];
            const float4* q = (const float4*)(arow + k0 + kka + 16);
            float4 f2 = q[0], f3 = q[1];
            a[0]  = (__bf16)f0.x; a[1]  = (__bf16)f0.y;
            a[2]  = (__bf16)f0.z; a[3]  = (__bf16)f0.w;
            a[4]  = (__bf16)f1.x; a[5]  = (__bf16)f1.y;
            a[6]  = (__bf16)f1.z; a[7]  = (__bf16)f1.w;
            a[8]  = (__bf16)f2.x; a[9]  = (__bf16)f2.y;
            a[10] = (__bf16)f2.z; a[11] = (__bf16)f2.w;
            a[12] = (__bf16)f3.x; a[13] = (__bf16)f3.y;
            a[14] = (__bf16)f3.z; a[15] = (__bf16)f3.w;
        }

        // B fragments: 32x16 bf16, lane n holds K {kkb..kkb+7, kkb+16..kkb+23}
        v16bf bf0, bf1;
        {
            v8bf lo = *(const v8bf*)(w0 + k0 + kkb);
            v8bf hi = *(const v8bf*)(w0 + k0 + kkb + 16);
            #pragma unroll
            for (int i = 0; i < 8; ++i) { bf0[i] = lo[i]; bf0[8 + i] = hi[i]; }
            lo = *(const v8bf*)(w1 + k0 + kkb);
            hi = *(const v8bf*)(w1 + k0 + kkb + 16);
            #pragma unroll
            for (int i = 0; i < 8; ++i) { bf1[i] = lo[i]; bf1[8 + i] = hi[i]; }
        }

        acc0 = __builtin_amdgcn_wmma_f32_16x16x32_bf16(false, a, false, bf0,
                                                       (short)0, acc0, false, false);
        acc1 = __builtin_amdgcn_wmma_f32_16x16x32_bf16(false, a, false, bf1,
                                                       (short)0, acc1, false, false);
    }

    // C/D layout: VGPR j -> row j (lanes 0-15) / row j+8 (lanes 16-31)
    {
        const int rbase = (lane < 16) ? 0 : 8;
        const int c0 = wave * 32 + (lane & 15);
        const int c1 = c0 + 16;
        const float bias0 = bias[c0], bias1 = bias[c1];
        #pragma unroll
        for (int j = 0; j < 8; ++j) {
            hbuf[rbase + j][c0] = acc0[j] + bias0;
            hbuf[rbase + j][c1] = acc1[j] + bias1;
        }
    }
    __syncthreads();

    // --- LayerNorm + L2-normalize phase ------------------------------------
    // 8 threads per row (contiguous lanes), 16 columns each.
    const int r = tid >> 3;
    const int g = tid & 7;

    float x[16];
    float s = 0.f, ss = 0.f;
    #pragma unroll
    for (int i = 0; i < 16; ++i) {
        float v = hbuf[r][g * 16 + i];
        x[i] = v; s += v; ss += v * v;
    }
    #pragma unroll
    for (int off = 1; off < 8; off <<= 1) {
        s  += __shfl_xor(s,  off, 32);
        ss += __shfl_xor(ss, off, 32);
    }
    const float mu   = s * (1.f / PN);
    const float var  = ss * (1.f / PN) - mu * mu;
    const float rstd = rsqrtf(var + LNEPS);

    float n2 = 0.f;
    #pragma unroll
    for (int i = 0; i < 16; ++i) {
        const int c = g * 16 + i;
        float v = (x[i] - mu) * rstd * gamma[c] + beta[c];
        x[i] = v; n2 += v * v;
    }
    #pragma unroll
    for (int off = 1; off < 8; off <<= 1) n2 += __shfl_xor(n2, off, 32);
    const float inv = 1.f / fmaxf(sqrtf(n2), 1e-12f);

    float* out = sem + ((size_t)b * LN_ + row0 + r) * PN + g * 16;
    #pragma unroll
    for (int i = 0; i < 16; ++i) out[i] = x[i] * inv;
}

// ---- kernel 2: per-edge weights (8 lanes per edge, shfl-reduced dot) -------
__global__ __launch_bounds__(256) void dgp_edges(const float* __restrict__ sem,
                                                 const float* __restrict__ coords,
                                                 const int*   __restrict__ eidx,
                                                 const float* __restrict__ alpha,
                                                 float*       __restrict__ wout) {
    const int t   = threadIdx.x;
    const int eid = blockIdx.x * 32 + (t >> 3);   // 32 edges / 256-thread block
    const int g   = t & 7;
    if (eid >= BN * EN) return;

    const int b = eid / EN;
    const int src = eidx[(size_t)eid * 2 + 0];
    const int dst = eidx[(size_t)eid * 2 + 1];

    const float* ps = sem + ((size_t)b * LN_ + src) * PN + g * 16;
    const float* pd = sem + ((size_t)b * LN_ + dst) * PN + g * 16;
    float dot = 0.f;
    #pragma unroll
    for (int i = 0; i < 4; ++i) {
        float4 va = ((const float4*)ps)[i];
        float4 vb = ((const float4*)pd)[i];
        dot += va.x * vb.x + va.y * vb.y + va.z * vb.z + va.w * vb.w;
    }
    #pragma unroll
    for (int off = 1; off < 8; off <<= 1) dot += __shfl_xor(dot, off, 32);

    if (g == 0) {
        const float* cs = coords + ((size_t)b * LN_ + src) * 2;
        const float* cd = coords + ((size_t)b * LN_ + dst) * 2;
        const float dx = cs[0] - cd[0], dy = cs[1] - cd[1];
        const float dist = sqrtf(dx * dx + dy * dy + 1e-8f);
        const float ssim = 1.f - dist * (1.f / 1.414f);
        const float a = 1.f / (1.f + expf(-alpha[0]));
        wout[eid] = a * dot + (1.f - a) * ssim;
    }
}

// ---------------------------------------------------------------------------
extern "C" void kernel_launch(void* const* d_in, const int* in_sizes, int n_in,
                              void* d_out, int out_size, void* d_ws, size_t ws_size,
                              hipStream_t stream) {
    const float* feat   = (const float*)d_in[0];   // (B,L,C)
    const float* coords = (const float*)d_in[1];   // (B,L,2)
    const int*   eidx   = (const int*)  d_in[2];   // (B,E,2)
    const float* alpha  = (const float*)d_in[3];   // scalar
    const float* Wp     = (const float*)d_in[4];   // (C,P)
    const float* bp     = (const float*)d_in[5];   // (P)
    const float* gamma  = (const float*)d_in[6];   // (P)
    const float* beta   = (const float*)d_in[7];   // (P)

    __bf16* Wt  = (__bf16*)d_ws;                          // 64 KB
    float*  sem = (float*)((char*)d_ws + 65536);          // 64 MB

    // Output: edge_index pass-through (B*E*2 int32 bits) then weights (B*E f32)
    hipMemcpyAsync(d_out, d_in[2], sizeof(int) * (size_t)BN * EN * 2,
                   hipMemcpyDeviceToDevice, stream);

    dgp_prep_w<<<(PN * CN + 255) / 256, 256, 0, stream>>>(Wp, Wt);

    dgp_proj_ln<<<BN * (LN_ / 16), 128, 0, stream>>>(feat, Wt, bp, gamma, beta, sem);

    float* wout = (float*)d_out + (size_t)BN * EN * 2;
    dgp_edges<<<(BN * EN + 31) / 32, 256, 0, stream>>>(sem, coords, eidx, alpha, wout);
}
